// RTGraphNet_54589034332984
// MI455X (gfx1250) — matrix operand order repaired and checked by reading, not compiled, not run on hardware
//
#include <hip/hip_runtime.h>
#include <hip/hip_bf16.h>

typedef __attribute__((ext_vector_type(16))) _Float16 v16h;
typedef __attribute__((ext_vector_type(8)))  _Float16 v8h;
typedef __attribute__((ext_vector_type(8)))  float    v8f;
typedef __attribute__((ext_vector_type(4)))  float    v4f;

#define GW   4      // waves per block in the WMMA GEMM
#define LROW 132    // padded LDS row stride (floats): kills half-wave bank conflicts

// ---------------- utility ----------------
__global__ void k_zero(float* __restrict__ p, int n) {
    int i = blockIdx.x * blockDim.x + threadIdx.x;
    if (i < n) p[i] = 0.0f;
}

// Per-node in-degree (same for both layers of a branch), then inverted in-place.
__global__ void k_count(const int* __restrict__ ei, float* __restrict__ cnt, int E) {
    int e = blockIdx.x * blockDim.x + threadIdx.x;
    if (e < E) atomicAdd(&cnt[ei[E + e]], 1.0f);
}
__global__ void k_inv(float* __restrict__ cnt, int N) {
    int n = blockIdx.x * blockDim.x + threadIdx.x;
    if (n < N) cnt[n] = 1.0f / fmaxf(cnt[n], 1.0f);
}

// Build Wcat = [EW0 | EW1 | EB | root] (K x 128, K padded to 32), f16,
// pre-swizzled into the WMMA B-operand lane layout:
//   index = tile*512 + lane*16 + e,  value = Wcat[k][col]
//   k = (lane<16 ? 0 : 8) + (e<8 ? e : e+8), col = tile*16 + (lane&15)
__global__ void k_prep_w(const float* __restrict__ ew, const float* __restrict__ eb,
                         const float* __restrict__ root, _Float16* __restrict__ Wh, int Din) {
    int tid = blockIdx.x * blockDim.x + threadIdx.x;   // 0..4095
    if (tid >= 4096) return;
    int t = tid >> 9;
    int l = (tid >> 4) & 31;
    int e = tid & 15;
    int k   = ((l < 16) ? 0 : 8) + ((e < 8) ? e : e + 8);
    int col = t * 16 + (l & 15);
    float v = 0.0f;
    if (k < Din) {
        int j = col & 31, grp = col >> 5;
        if (grp == 0)      v = ew[(k * 32 + j) * 2 + 0];   // EW0
        else if (grp == 1) v = ew[(k * 32 + j) * 2 + 1];   // EW1
        else if (grp == 2) v = eb[k * 32 + j];             // EB
        else               v = root[k * 32 + j];           // root
    }
    Wh[tid] = (_Float16)v;
}

// x (f32, [N,25]) -> f16 [N,32] zero-padded along K
__global__ void k_cvt_x(const float* __restrict__ X, _Float16* __restrict__ Xh, int N) {
    int tid = blockIdx.x * blockDim.x + threadIdx.x;
    int n = tid >> 5, k = tid & 31;
    if (n >= N) return;
    Xh[tid] = (k < 25) ? (_Float16)X[n * 25 + k] : (_Float16)0.0f;
}

// Y[N,128] = Xh[N,32] @ Wcat[32,128]; one wave per 16-row strip, K=32 -> 1 WMMA/tile.
// Epilogue: per-wave LDS transpose, then coalesced b128 row stores.
__global__ void k_gemm(const _Float16* __restrict__ Xh, const _Float16* __restrict__ Wh,
                       float* __restrict__ Y, int N) {
    __shared__ __align__(16) float tile[GW][16 * LROW];
    int lane = threadIdx.x & 31;
    int wave = threadIdx.x >> 5;
    int row0 = (blockIdx.x * GW + wave) * 16;
    if (row0 >= N) return;                       // uniform per wave; LDS region is per-wave

    // A operand: 16-bit A 16x32 layout. lanes 0-15: K=0..7,16..23 ; lanes 16-31: K=8..15,24..31
    int row = row0 + (lane & 15);
    int kb  = (lane < 16) ? 0 : 8;
    v16h a;
    if (row < N) {
        v8h lo = *(const v8h*)(Xh + row * 32 + kb);
        v8h hi = *(const v8h*)(Xh + row * 32 + 16 + kb);
#pragma unroll
        for (int i = 0; i < 8; ++i) { a[i] = lo[i]; a[i + 8] = hi[i]; }
    } else {
#pragma unroll
        for (int i = 0; i < 16; ++i) a[i] = (_Float16)0.0f;
    }

    float* lt = &tile[wave][0];
    int crow = (lane < 16) ? 0 : 8;              // C/D layout: VGPR r -> M = r (+8 for hi lanes)
    int nn   = lane & 15;
#pragma unroll
    for (int t = 0; t < 8; ++t) {
        v16h b = *(const v16h*)(Wh + t * 512 + lane * 16);
        v8f  c = {};
        c = __builtin_amdgcn_wmma_f32_16x16x32_f16(false, a, false, b, (short)0, c, false, false);
#pragma unroll
        for (int r = 0; r < 8; ++r)
            lt[(crow + r) * LROW + t * 16 + nn] = c[r];
    }
    __builtin_amdgcn_wave_barrier();             // same-wave LDS swizzle; DS cnt orders it

    // coalesced write-out: 16 rows x 128 floats; each lane stores 4 floats per row
#pragma unroll
    for (int r = 0; r < 16; ++r) {
        int gr = row0 + r;
        if (gr < N) {                            // wave-uniform guard
            v4f vv = *(const v4f*)(lt + r * LROW + lane * 4);
            *(v4f*)(Y + (size_t)gr * 128 + lane * 4) = vv;
        }
    }
}

// one wave per edge: msg = ea0*Y[src,0:32] + ea1*Y[src,32:64] + Y[src,64:96]
__global__ void k_edge(const int* __restrict__ ei, const float* __restrict__ ea,
                       const float* __restrict__ Y, float* __restrict__ S, int E) {
    int tid = blockIdx.x * blockDim.x + threadIdx.x;
    int e = tid >> 5, c = tid & 31;
    if (e >= E) return;
    int src = ei[e];
    int dst = ei[E + e];
    float a0 = ea[e * 2 + 0], a1 = ea[e * 2 + 1];
    const float* y = Y + (size_t)src * 128;
    float msg = a0 * y[c] + a1 * y[32 + c] + y[64 + c];
    atomicAdd(&S[(size_t)dst * 32 + c], msg);
}

// h = relu(Y[:,96:128] + S*invcnt + bias) -> f16 for next GEMM
__global__ void k_fin1(const float* __restrict__ Y, const float* __restrict__ S,
                       const float* __restrict__ invc, const float* __restrict__ bias,
                       _Float16* __restrict__ Hh, int N) {
    int tid = blockIdx.x * blockDim.x + threadIdx.x;
    int n = tid >> 5, c = tid & 31;
    if (n >= N) return;
    float h = Y[(size_t)n * 128 + 96 + c] + S[(size_t)n * 32 + c] * invc[n] + bias[c];
    Hh[(size_t)n * 32 + c] = (_Float16)fmaxf(h, 0.0f);
}

// layer-2 finalize fused with global mean-pool accumulation (LDS then global atomics)
__global__ void k_fin2(const float* __restrict__ Y, const float* __restrict__ S,
                       const float* __restrict__ invc, const float* __restrict__ bias,
                       float* __restrict__ emb_acc, int N) {
    __shared__ float sacc[32];
    int tid  = threadIdx.x;
    int gtid = blockIdx.x * blockDim.x + tid;
    int n = gtid >> 5, c = tid & 31;
    if (tid < 32) sacc[tid] = 0.0f;
    __syncthreads();
    float h = 0.0f;
    if (n < N)
        h = fmaxf(Y[(size_t)n * 128 + 96 + c] + S[(size_t)n * 32 + c] * invc[n] + bias[c], 0.0f);
    atomicAdd(&sacc[c], h);            // ds_add_f32
    __syncthreads();
    if (tid < 32) atomicAdd(&emb_acc[tid], sacc[tid]);
}

__global__ void k_head(const float* __restrict__ emb_acc,
                       const float* __restrict__ cw, const float* __restrict__ cb,
                       const float* __restrict__ rw, const float* __restrict__ rb,
                       float invNf, float invNs, float* __restrict__ out) {
    __shared__ float se[64], sp[64];
    int c = threadIdx.x;               // 0..63
    float e = emb_acc[c] * (c < 32 ? invNf : invNs);
    out[2 + c] = e;
    se[c] = e * cw[c];
    sp[c] = e * rw[c];
    __syncthreads();
    for (int s = 32; s > 0; s >>= 1) {
        if (c < s) { se[c] += se[c + s]; sp[c] += sp[c + s]; }
        __syncthreads();
    }
    if (c == 0) { out[0] = se[0] + cb[0]; out[1] = sp[0] + rb[0]; }
}

// ---------------- host ----------------
extern "C" void kernel_launch(void* const* d_in, const int* in_sizes, int n_in,
                              void* d_out, int out_size, void* d_ws, size_t ws_size,
                              hipStream_t stream) {
    (void)n_in; (void)out_size; (void)ws_size;
    const float* x_full  = (const float*)d_in[0];
    const int*   ei_full = (const int*)  d_in[1];
    const float* ea_full = (const float*)d_in[2];
    const float* x_site  = (const float*)d_in[3];
    const int*   ei_site = (const int*)  d_in[4];
    const float* ea_site = (const float*)d_in[5];

    int Nf = in_sizes[0] / 25, Ef = in_sizes[2] / 2;
    int Ns = in_sizes[3] / 25, Es = in_sizes[5] / 2;
    int Nmax = (Nf > Ns) ? Nf : Ns;

    auto al = [](size_t v) { return (v + 255) & ~(size_t)255; };
    char* ws = (char*)d_ws;
    size_t off = 0;
    float*     Y   = (float*)(ws + off);     off += al((size_t)Nmax * 128 * 4);
    float*     S   = (float*)(ws + off);     off += al((size_t)Nmax * 32 * 4);
    float*     cnt = (float*)(ws + off);     off += al((size_t)Nmax * 4);
    _Float16*  Xh  = (_Float16*)(ws + off);  off += al((size_t)Nmax * 32 * 2);
    _Float16*  Hh  = (_Float16*)(ws + off);  off += al((size_t)Nmax * 32 * 2);
    _Float16*  Wh  = (_Float16*)(ws + off);  off += al((size_t)4096 * 2);
    float*     emb = (float*)(ws + off);     off += 256;

    const int T = 256;
    auto run_branch = [&](const float* X, const int* EI, const float* EA,
                          const float* w1ew, const float* w1eb, const float* w1r, const float* w1b,
                          const float* w2ew, const float* w2eb, const float* w2r, const float* w2b,
                          int N, int E, float* emb_out) {
        int gNC = (N * 32 + T - 1) / T;
        int gN  = (N + T - 1) / T;
        int gG  = (N + GW * 16 - 1) / (GW * 16);
        int gE  = (int)(((size_t)E * 32 + T - 1) / T);
        int gEc = (E + T - 1) / T;
        // degree (shared by both layers)
        k_zero  <<<gN,  T, 0, stream>>>(cnt, N);
        k_count <<<gEc, T, 0, stream>>>(EI, cnt, E);
        k_inv   <<<gN,  T, 0, stream>>>(cnt, N);
        // layer 1 (K = 25 padded to 32)
        k_prep_w<<<16, T, 0, stream>>>(w1ew, w1eb, w1r, Wh, 25);
        k_cvt_x <<<gNC, T, 0, stream>>>(X, Xh, N);
        k_zero  <<<gNC, T, 0, stream>>>(S, N * 32);
        k_gemm  <<<gG, GW * 32, 0, stream>>>(Xh, Wh, Y, N);
        k_edge  <<<gE,  T, 0, stream>>>(EI, EA, Y, S, E);
        k_fin1  <<<gNC, T, 0, stream>>>(Y, S, cnt, w1b, Hh, N);
        // layer 2 (K = 32)
        k_prep_w<<<16, T, 0, stream>>>(w2ew, w2eb, w2r, Wh, 32);
        k_zero  <<<gNC, T, 0, stream>>>(S, N * 32);
        k_gemm  <<<gG, GW * 32, 0, stream>>>(Hh, Wh, Y, N);
        k_edge  <<<gE,  T, 0, stream>>>(EI, EA, Y, S, E);
        k_fin2  <<<gNC, T, 0, stream>>>(Y, S, cnt, w2b, emb_out, N);
    };

    k_zero<<<1, 64, 0, stream>>>(emb, 64);
    run_branch(x_full, ei_full, ea_full,
               (const float*)d_in[6],  (const float*)d_in[7],  (const float*)d_in[8],  (const float*)d_in[9],
               (const float*)d_in[10], (const float*)d_in[11], (const float*)d_in[12], (const float*)d_in[13],
               Nf, Ef, emb);
    run_branch(x_site, ei_site, ea_site,
               (const float*)d_in[14], (const float*)d_in[15], (const float*)d_in[16], (const float*)d_in[17],
               (const float*)d_in[18], (const float*)d_in[19], (const float*)d_in[20], (const float*)d_in[21],
               Ns, Es, emb + 32);
    k_head<<<1, 64, 0, stream>>>(emb,
               (const float*)d_in[22], (const float*)d_in[23],
               (const float*)d_in[24], (const float*)d_in[25],
               1.0f / (float)Nf, 1.0f / (float)Ns, (float*)d_out);
}
